// CompresSAE_15006615734287
// MI455X (gfx1250) — matrix operand order repaired and checked by reading, not compiled, not run on hardware
//
#include <hip/hip_runtime.h>
#include <hip/hip_bf16.h>

typedef __bf16 v16bf __attribute__((ext_vector_type(16)));
typedef float  v8f   __attribute__((ext_vector_type(8)));

#define D_DIM 768
#define E_DIM 16384
#define B_ROWS 8192
#define K_TOP 64

// fused encode tiling: 32 rows x 32 cols per E-tile, 8 waves = 2(row) x 2(col) x 2(K-split)
#define BM 32
#define BN 32
#define KSPLIT_LEN 384            // 768 / 2
#define KSTEPS 12                 // 384 / 32
#define LDA 776                   // padded bf16 row stride (388 dwords, %64 = 4 -> conflict-free)
#define LDB 776
#define NTHREADS 256

// dynamic LDS layout (bytes)
#define SA_BYTES   (BM * LDA * 2)            // 49664
#define SB_BYTES   (BN * LDB * 2)            // 49664
#define SACC_BYTES (2 * BM * BN * 4)         // 8192 (two K-split partials)
#define SVAL_BYTES (BM * K_TOP * 4)          // 8192
#define SIDX_BYTES (BM * K_TOP * 4)          // 8192
#define SMISC_BYTES (3 * BM * 4)
#define SMEM_TOTAL (SA_BYTES + SB_BYTES + SACC_BYTES + SVAL_BYTES + SIDX_BYTES + SMISC_BYTES) // 124288

// ---- CDNA5 async memory->LDS copy (ASYNCcnt path, ISA 08_async_tensor) -----
__device__ __forceinline__ void async_b128(uint32_t ldsAddr, const void* gaddr) {
    asm volatile("global_load_async_to_lds_b128 %0, %1, off"
                 :: "v"(ldsAddr), "v"(gaddr) : "memory");
}
__device__ __forceinline__ void wait_async0() {
    asm volatile("s_wait_asynccnt 0x0" ::: "memory");
}

// ---- WMMA fragment loads from LDS (ISA 7.12.2 layouts) ----------------------
__device__ __forceinline__ v16bf load_a_frag(const __bf16* p) {
    v16bf f;
    ((uint4*)&f)[0] = *(const uint4*)p;          // K: k0+8h   .. +7
    ((uint4*)&f)[1] = *(const uint4*)(p + 16);   // K: k0+16+8h.. +7
    return f;
}
__device__ __forceinline__ v16bf load_b_frag(const __bf16* p) {
    v16bf f;
    ((uint4*)&f)[0] = *(const uint4*)p;          // K: k0+8h   .. +7
    ((uint4*)&f)[1] = *(const uint4*)(p + 16);   // K: k0+16+8h.. +7
    return f;
}

// ---------------------------------------------------------------------------
// Kernel 0: transpose + convert encoder_w [D, E] f32 -> wt [E, D] bf16
// ---------------------------------------------------------------------------
__global__ __launch_bounds__(256)
void enc_transpose(const float* __restrict__ w, __bf16* __restrict__ wt) {
    __shared__ float tile[32][33];
    const int eBase = blockIdx.x * 32;
    const int dBase = blockIdx.y * 32;
    const int tx = threadIdx.x;   // 0..31
    const int ty = threadIdx.y;   // 0..7
#pragma unroll
    for (int i = 0; i < 32; i += 8)
        tile[ty + i][tx] = w[(size_t)(dBase + ty + i) * E_DIM + eBase + tx];
    __syncthreads();
#pragma unroll
    for (int i = 0; i < 32; i += 8)
        wt[(size_t)(eBase + ty + i) * D_DIM + dBase + tx] = (__bf16)tile[tx][ty + i];
}

// ---------------------------------------------------------------------------
// Kernel 1: xn[b,:] = x[b,:] / ||x[b,:]||  -> bf16
// ---------------------------------------------------------------------------
__global__ __launch_bounds__(256)
void normalize_rows(const float* __restrict__ x, __bf16* __restrict__ xn) {
    const int row = blockIdx.x;
    const int t = threadIdx.x;
    const float* xr = x + (size_t)row * D_DIM;
    float v0 = xr[t], v1 = xr[t + 256], v2 = xr[t + 512];
    float s = v0 * v0 + v1 * v1 + v2 * v2;
#pragma unroll
    for (int off = 16; off > 0; off >>= 1) s += __shfl_down(s, off, 32);
    __shared__ float red[8];
    __shared__ float rn;
    if ((t & 31) == 0) red[t >> 5] = s;
    __syncthreads();
    if (t == 0) {
        float tot = 0.f;
#pragma unroll
        for (int i = 0; i < 8; ++i) tot += red[i];
        rn = __builtin_amdgcn_rsqf(tot);
    }
    __syncthreads();
    const float r = rn;
    __bf16* d = xn + (size_t)row * D_DIM;
    d[t]       = (__bf16)(v0 * r);
    d[t + 256] = (__bf16)(v1 * r);
    d[t + 512] = (__bf16)(v2 * r);
}

// ---------------------------------------------------------------------------
// Kernel 2: fused  e_pre = xn @ W_e + b  with streaming per-row top-64.
// e_pre ([B,E] = 512 MB fp32) never touches global memory.
// ---------------------------------------------------------------------------
__global__ __launch_bounds__(NTHREADS, 1)
void sae_encode_topk(const __bf16* __restrict__ xn,
                     const __bf16* __restrict__ wt,
                     const float* __restrict__ bias,
                     float* __restrict__ topValG,
                     int* __restrict__ topIdxG) {
    extern __shared__ char smem[];
    __bf16* sA   = (__bf16*)smem;
    __bf16* sB   = (__bf16*)(smem + SA_BYTES);
    float*  sAcc = (float*)(smem + SA_BYTES + SB_BYTES);
    float*  sVal = (float*)(smem + SA_BYTES + SB_BYTES + SACC_BYTES);
    int*    sIdx = (int*)(smem + SA_BYTES + SB_BYTES + SACC_BYTES + SVAL_BYTES);
    float*  sMin = (float*)(smem + SA_BYTES + SB_BYTES + SACC_BYTES + SVAL_BYTES + SIDX_BYTES);
    int*    sMinPos = (int*)((char*)sMin + BM * 4);
    int*    sCnt    = (int*)((char*)sMin + 2 * BM * 4);
    const uint32_t sA_lds = (uint32_t)(uintptr_t)sA;
    const uint32_t sB_lds = (uint32_t)(uintptr_t)sB;

    const int t    = threadIdx.x;
    const int lane = t & 31;
    const int wave = t >> 5;            // 0..7
    const int wc   = wave & 1;          // col tile
    const int wr   = (wave >> 1) & 1;   // row tile
    const int ks   = wave >> 2;         // K-split half
    const int half = lane >> 4;
    const int l16  = lane & 15;
    const int rowBase = blockIdx.x * BM;

    // async-stage A (32 rows x 768 bf16): 8 threads/row, 192 B each
    {
        const int r = t >> 3, seg = t & 7;
        const __bf16* src = xn + (size_t)(rowBase + r) * D_DIM + seg * 96;
        uint32_t dst = sA_lds + (uint32_t)(r * LDA + seg * 96) * 2u;
#pragma unroll
        for (int u = 0; u < 12; ++u) async_b128(dst + u * 16u, src + u * 8);
    }
    if (t < BM) { sCnt[t] = 0; sMin[t] = __builtin_inff(); sMinPos[t] = 0; }

    const int aOff = (wr * 16 + l16) * LDA + ks * KSPLIT_LEN + half * 8;
    const int bOff = (wc * 16 + l16) * LDB + ks * KSPLIT_LEN + half * 8;

    for (int tile = 0; tile < E_DIM / BN; ++tile) {
        const int colBase = tile * BN;
        __syncthreads();   // previous top-k done with sAcc; previous compute done with sB
        // async-stage B tile (32 cols x 768 bf16)
        {
            const int c = t >> 3, seg = t & 7;
            const __bf16* src = wt + (size_t)(colBase + c) * D_DIM + seg * 96;
            uint32_t dst = sB_lds + (uint32_t)(c * LDB + seg * 96) * 2u;
#pragma unroll
            for (int u = 0; u < 12; ++u) async_b128(dst + u * 16u, src + u * 8);
        }
        wait_async0();
        __syncthreads();

        // 16x16 f32 tile over this wave's K half, software-pipelined
        v8f acc = {0.f, 0.f, 0.f, 0.f, 0.f, 0.f, 0.f, 0.f};
        v16bf a0 = load_a_frag(sA + aOff);
        v16bf b0 = load_b_frag(sB + bOff);
#pragma unroll
        for (int step = 0; step < KSTEPS; ++step) {
            v16bf a1, b1;
            if (step < KSTEPS - 1) {
                a1 = load_a_frag(sA + aOff + (step + 1) * 32);
                b1 = load_b_frag(sB + bOff + (step + 1) * 32);
            } else { a1 = a0; b1 = b0; }
            acc = __builtin_amdgcn_wmma_f32_16x16x32_bf16(false, a0, false, b0,
                                                          (short)0, acc, false, false);
            a0 = a1; b0 = b1;
        }
        // spill partial sums (C layout: VGPR v -> M = v + 8*half, N = l16)
        {
            const int mrow = wr * 16 + half * 8;
            const int ncol = wc * 16 + l16;
            float* dstAcc = sAcc + ks * BM * BN;
#pragma unroll
            for (int v = 0; v < 8; ++v) dstAcc[(mrow + v) * BN + ncol] = acc[v];
        }
        __syncthreads();

        // streaming top-64 by |value|; each wave owns 4 rows
        const float biasL = bias[colBase + lane];
        for (int rr = 0; rr < 4; ++rr) {
            const int r = wave * 4 + rr;
            const float val = sAcc[r * BN + lane] + sAcc[BM * BN + r * BN + lane] + biasL;
            const float av  = __builtin_fabsf(val);
            const int gcol  = colBase + lane;
            bool done = false;
            while (true) {
                const bool pred = (!done) && ((sCnt[r] < K_TOP) || (av > sMin[r]));
                const unsigned m = __builtin_amdgcn_ballot_w32(pred);
                if (m == 0u) break;
                const int s = __builtin_ctz(m);
                const float vs = __shfl(val, s, 32);
                const float as = __shfl(av, s, 32);
                const int   cs = __shfl(gcol, s, 32);
                if (lane == s) done = true;
                const int cnt = sCnt[r];
                if (cnt < K_TOP) {
                    if (lane == 0) {
                        sVal[r * K_TOP + cnt] = vs;
                        sIdx[r * K_TOP + cnt] = cs;
                        if (as < sMin[r]) { sMin[r] = as; sMinPos[r] = cnt; }
                        sCnt[r] = cnt + 1;
                    }
                } else {
                    if (as <= sMin[r]) continue;       // stale candidate (uniform)
                    const int mp = sMinPos[r];
                    if (lane == 0) { sVal[r * K_TOP + mp] = vs; sIdx[r * K_TOP + mp] = cs; }
                    // wave-parallel rescan of the 64 entries (patch replaced slot)
                    float e0 = (lane == mp)      ? as : __builtin_fabsf(sVal[r * K_TOP + lane]);
                    float e1 = (lane + 32 == mp) ? as : __builtin_fabsf(sVal[r * K_TOP + lane + 32]);
                    float mv; int mi;
                    if (e0 < e1) { mv = e0; mi = lane; } else { mv = e1; mi = lane + 32; }
#pragma unroll
                    for (int off = 16; off > 0; off >>= 1) {
                        const float ov = __shfl_down(mv, off, 32);
                        const int   oi = __shfl_down(mi, off, 32);
                        if (ov < mv) { mv = ov; mi = oi; }
                    }
                    if (lane == 0) { sMin[r] = mv; sMinPos[r] = mi; }
                }
            }
        }
    }
    __syncthreads();
    for (int i = t; i < BM * K_TOP; i += NTHREADS) {
        const int r = i >> 6, j = i & 63;
        topValG[(size_t)(rowBase + r) * K_TOP + j] = sVal[r * K_TOP + j];
        topIdxG[(size_t)(rowBase + r) * K_TOP + j] = sIdx[r * K_TOP + j];
    }
}

// ---------------------------------------------------------------------------
// Kernel 3: sparse decoder  out[b,:] = sum_t val[b,t] * W_d[idx[b,t], :]
// ---------------------------------------------------------------------------
__global__ __launch_bounds__(256)
void sae_decode(const float* __restrict__ topValG, const int* __restrict__ topIdxG,
                const float* __restrict__ dec, float* __restrict__ out) {
    __shared__ float v[K_TOP];
    __shared__ int   id[K_TOP];
    const int row = blockIdx.x;
    const int t = threadIdx.x;
    if (t < K_TOP) {
        v[t]  = topValG[(size_t)row * K_TOP + t];
        id[t] = topIdxG[(size_t)row * K_TOP + t];
    }
    __syncthreads();
    float a0 = 0.f, a1 = 0.f, a2 = 0.f;
#pragma unroll 4
    for (int k = 0; k < K_TOP; ++k) {
        const float* dr = dec + (size_t)id[k] * D_DIM;
        const float s = v[k];
        a0 = __builtin_fmaf(s, dr[t],       a0);
        a1 = __builtin_fmaf(s, dr[t + 256], a1);
        a2 = __builtin_fmaf(s, dr[t + 512], a2);
    }
    float* o = out + (size_t)row * D_DIM;
    o[t] = a0; o[t + 256] = a1; o[t + 512] = a2;
}

// ---------------------------------------------------------------------------
extern "C" void kernel_launch(void* const* d_in, const int* in_sizes, int n_in,
                              void* d_out, int out_size, void* d_ws, size_t ws_size,
                              hipStream_t stream) {
    (void)in_sizes; (void)n_in; (void)out_size; (void)ws_size;
    const float* x  = (const float*)d_in[0];   // [B, D]
    const float* ew = (const float*)d_in[1];   // [D, E]
    const float* eb = (const float*)d_in[2];   // [E]
    const float* dw = (const float*)d_in[3];   // [E, D]
    // d_in[4] is k == 64 (hardcoded as K_TOP)
    float* out = (float*)d_out;

    // workspace layout (~42 MB)
    char* ws = (char*)d_ws;
    __bf16* xn = (__bf16*)ws;                                               // B*D bf16
    __bf16* wt = (__bf16*)(ws + (size_t)B_ROWS * D_DIM * 2);                // E*D bf16
    float*  tv = (float*)(ws + (size_t)B_ROWS * D_DIM * 2
                             + (size_t)E_DIM * D_DIM * 2);                  // B*64 f32
    int*    ti = (int*)((char*)tv + (size_t)B_ROWS * K_TOP * 4);            // B*64 i32

    enc_transpose<<<dim3(E_DIM / 32, D_DIM / 32), dim3(32, 8), 0, stream>>>(ew, wt);
    normalize_rows<<<B_ROWS, 256, 0, stream>>>(x, xn);
    sae_encode_topk<<<B_ROWS / BM, NTHREADS, SMEM_TOTAL, stream>>>(xn, wt, eb, tv, ti);
    sae_decode<<<B_ROWS, 256, 0, stream>>>(tv, ti, dw, out);
}
